// ContextPeftAdaptorLora_6760278524564
// MI455X (gfx1250) — compile-verified, matrix operand order checked by reading, not run on hardware
//
#include <hip/hip_runtime.h>

// ---------------------------------------------------------------------------
// Fused LoRA forward for gfx1250 (MI455X), bf16 WMMA with fp32 accumulation.
//   out = x @ W^T + b + sum_n mask_n * SCALE * (x @ A_n^T) @ Bw_n^T
// Reformulated as one GEMM with K = 2048 plus a 64-wide "extra-K" epilogue:
//   Hm[m, n*16+r] = SCALE * mask[n,m] * (x @ A^T)      (precomputed, bf16)
//   out = [x | Hm] @ [W | Bwcat]^T + b
// CDNA5 specifics: v_wmma_f32_16x16x32_bf16, async-to-LDS DMA for the bf16
// W tiles (ASYNCcnt + s_wait_asynccnt), global_prefetch_b8 for upcoming x.
// ---------------------------------------------------------------------------

typedef __bf16 bf16_t;
typedef bf16_t v16bf __attribute__((ext_vector_type(16)));
typedef bf16_t v8bf  __attribute__((ext_vector_type(8)));
typedef bf16_t v4bf  __attribute__((ext_vector_type(4)));
typedef float  v8f   __attribute__((ext_vector_type(8)));

#define WMMA_BF16(a, b, c) \
  __builtin_amdgcn_wmma_f32_16x16x32_bf16(false, (a), false, (b), (short)0, (c), false, false)

static constexpr int M_TOT  = 16384;  // B*S
static constexpr int K_IN   = 2048;   // D_IN
static constexpr int N_OUT  = 2048;   // D_OUT
static constexpr int KE     = 64;     // N_ADP * R
static constexpr float LORA_SCALE = 2.0f;  // 32/16

__device__ __forceinline__ v16bf cat16(v8bf lo, v8bf hi) {
  return __builtin_shufflevector(lo, hi, 0,1,2,3,4,5,6,7,8,9,10,11,12,13,14,15);
}
__device__ __forceinline__ v4bf cvt4(float4 f) {
  v4bf r; r[0]=(bf16_t)f.x; r[1]=(bf16_t)f.y; r[2]=(bf16_t)f.z; r[3]=(bf16_t)f.w; return r;
}
__device__ __forceinline__ v8bf cvt8(float4 a, float4 b) {
  v8bf r;
  r[0]=(bf16_t)a.x; r[1]=(bf16_t)a.y; r[2]=(bf16_t)a.z; r[3]=(bf16_t)a.w;
  r[4]=(bf16_t)b.x; r[5]=(bf16_t)b.y; r[6]=(bf16_t)b.z; r[7]=(bf16_t)b.w;
  return r;
}

// Async DMA: global (bf16, 16B/lane) -> LDS, tracked by ASYNCcnt.
__device__ __forceinline__ void async_copy_b128(const bf16_t* g, void* lds_generic) {
  unsigned lds = (unsigned)(uintptr_t)lds_generic;   // low 32 bits == LDS offset
  asm volatile("global_load_async_to_lds_b128 %0, %1, off"
               :: "v"(lds), "v"(g) : "memory");
}
__device__ __forceinline__ void wait_async0() {
  asm volatile("s_wait_asynccnt 0x0" ::: "memory");
}

// ---------------------------------------------------------------------------
// Kernel P: dtype prep. W -> bf16 (same layout), A -> bf16 (same flat layout,
// [64 x 2048]), Bw[n,o,r] -> Bb[o][n*16+r] bf16 (the 64 extra K-columns of W).
// ---------------------------------------------------------------------------
__global__ __launch_bounds__(256) void lora_prep_kernel(
    const float* __restrict__ W, const float* __restrict__ A,
    const float* __restrict__ Bw,
    bf16_t* __restrict__ Wb, bf16_t* __restrict__ Ab, bf16_t* __restrict__ Bb)
{
  const int idx = blockIdx.x * 256 + threadIdx.x;      // 0 .. 1048575
  {  // W: 4M elements, 4 per thread, vectorized
    float4 f = *(const float4*)(W + (size_t)idx * 4);
    *(v4bf*)(Wb + (size_t)idx * 4) = cvt4(f);
  }
  if (idx < 32768) {  // A: 64*2048 elements, contiguous
    float4 f = *(const float4*)(A + (size_t)idx * 4);
    *(v4bf*)(Ab + (size_t)idx * 4) = cvt4(f);
  }
  if (idx < 131072) { // Bb[o*64 + n*16 + r] = Bw[n*2048*16 + o*16 + r]
    int o = idx >> 6, c = idx & 63, n = c >> 4, r = c & 15;
    Bb[idx] = (bf16_t)Bw[(size_t)n * 32768 + (size_t)o * 16 + r];
  }
}

// ---------------------------------------------------------------------------
// Kernel H: Hm[m, 0..63] = SCALE * mask[n,m] * (x @ A^T).  One wave per
// 16-row strip; 4 WMMA per K-stage (64 output cols).  B operand = rows of
// bf16 A read directly with the 16-bit B-operand K-interleave.
// ---------------------------------------------------------------------------
__global__ __launch_bounds__(256) void lora_h_kernel(
    const float* __restrict__ X, const bf16_t* __restrict__ Ab,
    const float* __restrict__ mask, bf16_t* __restrict__ Hm)
{
  const int tid  = threadIdx.x;
  const int lane = tid & 31;
  const int wave = tid >> 5;
  const int m0   = (blockIdx.x * 8 + wave) * 16;   // grid = 128 blocks
  const int l15  = lane & 15;
  const int kb   = (lane < 16) ? 0 : 8;            // K-interleave per half-wave

  v8f acc[4] = { {}, {}, {}, {} };
  const float* xrow = X + (size_t)(m0 + l15) * K_IN;

  for (int ks = 0; ks < K_IN / 32; ++ks) {
    const int k0 = ks * 32 + kb;
    // A operand (16x32 bf16): lane half gets K {k0..k0+7, k0+16..k0+23}
    float4 f0 = *(const float4*)(xrow + k0);
    float4 f1 = *(const float4*)(xrow + k0 + 4);
    float4 f2 = *(const float4*)(xrow + k0 + 16);
    float4 f3 = *(const float4*)(xrow + k0 + 20);
    v16bf a = cat16(cvt8(f0, f1), cvt8(f2, f3));
#pragma unroll
    for (int t = 0; t < 4; ++t) {
      const bf16_t* ap = Ab + (size_t)(t * 16 + l15) * K_IN + k0;
      v16bf b = cat16(*(const v8bf*)ap, *(const v8bf*)(ap + 16));
      acc[t] = WMMA_BF16(a, b, acc[t]);
    }
  }

  // Epilogue: fold SCALE * mask[n, m] per row, store bf16.
  const int rbase = m0 + ((lane >> 4) << 3);
#pragma unroll
  for (int t = 0; t < 4; ++t) {        // t == adaptor index (16 cols each)
#pragma unroll
    for (int j = 0; j < 8; ++j) {
      const int row = rbase + j;
      float v = acc[t][j] * LORA_SCALE * mask[(size_t)t * M_TOT + row];
      Hm[(size_t)row * KE + t * 16 + l15] = (bf16_t)v;
    }
  }
}

// ---------------------------------------------------------------------------
// Kernel G: main GEMM.  128x128 block tile, 8 waves of 64x32, double-buffered
// LDS.  x tile: fp32 global loads + VALU cvt to bf16 (must touch VGPRs).
// W tile: bf16 async DMA straight to LDS (no VGPR staging).  8 WMMA/stage.
// Epilogue: 2 more WMMA K-steps over [Hm | Bb] + bias, store fp32.
// ---------------------------------------------------------------------------
__global__ __launch_bounds__(256) void lora_main_gemm(
    const float* __restrict__ X, const bf16_t* __restrict__ Wb,
    const bf16_t* __restrict__ Hm, const bf16_t* __restrict__ Bb,
    const float* __restrict__ bias, float* __restrict__ out)
{
  __shared__ __align__(16) bf16_t Xs[2][128][40];   // 32 K + 8 pad
  __shared__ __align__(16) bf16_t Ws[2][128][40];

  const int tid  = threadIdx.x;
  const int lane = tid & 31;
  const int wave = tid >> 5;
  const int wm   = wave & 1;          // 2 waves over M (64 rows each)
  const int wn   = wave >> 1;         // 4 waves over N (32 cols each)
  const int n0   = blockIdx.x * 128;
  const int m0   = blockIdx.y * 128;
  const int l15  = lane & 15;
  const int kb   = (lane < 16) ? 0 : 8;

  v8f acc[4][2];
#pragma unroll
  for (int mt = 0; mt < 4; ++mt)
#pragma unroll
    for (int nt = 0; nt < 2; ++nt) acc[mt][nt] = (v8f){};

  auto loadX = [&](int ks, float4 r[4]) {
    const float* xb = X + (size_t)m0 * K_IN + ks * 32;
#pragma unroll
    for (int i = 0; i < 4; ++i) {
      int fl = tid + i * 256, row = fl >> 3, ch = fl & 7;
      r[i] = *(const float4*)(xb + (size_t)row * K_IN + ch * 4);
    }
  };
  auto storeX = [&](int buf, float4 r[4]) {
#pragma unroll
    for (int i = 0; i < 4; ++i) {
      int fl = tid + i * 256, row = fl >> 3, ch = fl & 7;
      *(v4bf*)&Xs[buf][row][ch * 4] = cvt4(r[i]);
    }
  };
  // W tile: 128 rows x 32 K of bf16 == 8 KB; 512 x b128 chunks; 2 per thread.
  auto asyncW = [&](int ks, int buf) {
    const bf16_t* wb = Wb + (size_t)n0 * K_IN + ks * 32;
#pragma unroll
    for (int i = 0; i < 2; ++i) {
      int fl = tid + i * 256, row = fl >> 2, ch = fl & 3;
      async_copy_b128(wb + (size_t)row * K_IN + ch * 8, &Ws[buf][row][ch * 8]);
    }
  };
  auto compute = [&](int buf) {
    v16bf a[4], b[2];
#pragma unroll
    for (int mt = 0; mt < 4; ++mt) {
      const bf16_t* p = &Xs[buf][wm * 64 + mt * 16 + l15][0];
      a[mt] = cat16(*(const v8bf*)(p + kb), *(const v8bf*)(p + kb + 16));
    }
#pragma unroll
    for (int nt = 0; nt < 2; ++nt) {
      const bf16_t* p = &Ws[buf][wn * 32 + nt * 16 + l15][0];
      b[nt] = cat16(*(const v8bf*)(p + kb), *(const v8bf*)(p + kb + 16));
    }
#pragma unroll
    for (int mt = 0; mt < 4; ++mt)
#pragma unroll
      for (int nt = 0; nt < 2; ++nt)
        acc[mt][nt] = WMMA_BF16(a[mt], b[nt], acc[mt][nt]);
  };

  // --- prologue: stage 0 (async W DMA + reg-staged x) ---
  {
    asyncW(0, 0);
    float4 rx[4]; loadX(0, rx); storeX(0, rx);
    wait_async0();
  }

  // --- double-buffered main K loop (64 stages of K=32) ---
  constexpr int NSTAGE = K_IN / 32;
  for (int ks = 0; ks < NSTAGE; ++ks) {
    __syncthreads();                        // publish stage ks to all waves
    float4 rx[4];
    if (ks + 1 < NSTAGE) {
      asyncW(ks + 1, (ks + 1) & 1);         // DMA overlaps this stage's WMMA
      loadX(ks + 1, rx);                    // fp32 loads overlap WMMA
      if (ks + 2 < NSTAGE)                  // gfx1250 global_prefetch_b8
        __builtin_prefetch(X + (size_t)(m0 + (tid >> 3)) * K_IN + (ks + 2) * 32, 0, 3);
    }
    compute(ks & 1);
    __syncthreads();                        // all waves done reading buffers
    if (ks + 1 < NSTAGE) {
      storeX((ks + 1) & 1, rx);
      wait_async0();                        // our W DMA landed (barrier next)
    }
  }

  // --- LoRA epilogue: 2 extra WMMA K-steps over [Hm | Bwcat] ---
#pragma unroll
  for (int kt = 0; kt < 2; ++kt) {
    const int kb2 = kt * 32 + kb;
    v16bf a[4], b[2];
#pragma unroll
    for (int mt = 0; mt < 4; ++mt) {
      const bf16_t* p = Hm + (size_t)(m0 + wm * 64 + mt * 16 + l15) * KE + kb2;
      a[mt] = cat16(*(const v8bf*)p, *(const v8bf*)(p + 16));
    }
#pragma unroll
    for (int nt = 0; nt < 2; ++nt) {
      const bf16_t* p = Bb + (size_t)(n0 + wn * 32 + nt * 16 + l15) * KE + kb2;
      b[nt] = cat16(*(const v8bf*)p, *(const v8bf*)(p + 16));
    }
#pragma unroll
    for (int mt = 0; mt < 4; ++mt)
#pragma unroll
      for (int nt = 0; nt < 2; ++nt)
        acc[mt][nt] = WMMA_BF16(a[mt], b[nt], acc[mt][nt]);
  }

  // --- bias + store (C/D layout: VGPR j -> row j (+8 for hi lanes), N=lane) ---
  const int rbase = m0 + wm * 64 + ((lane >> 4) << 3);
#pragma unroll
  for (int nt = 0; nt < 2; ++nt) {
    const int col = n0 + wn * 32 + nt * 16 + l15;
    const float bv = bias[col];
#pragma unroll
    for (int mt = 0; mt < 4; ++mt) {
#pragma unroll
      for (int j = 0; j < 8; ++j) {
        const int row = rbase + mt * 16 + j;
        out[(size_t)row * N_OUT + col] = acc[mt][nt][j] + bv;
      }
    }
  }
}

// ---------------------------------------------------------------------------
extern "C" void kernel_launch(void* const* d_in, const int* in_sizes, int n_in,
                              void* d_out, int out_size, void* d_ws, size_t ws_size,
                              hipStream_t stream) {
  const float* x    = (const float*)d_in[0];   // [4,4096,2048]
  const float* mask = (const float*)d_in[1];   // [4,4,4096,1]
  const float* W    = (const float*)d_in[2];   // [2048,2048]
  const float* bias = (const float*)d_in[3];   // [2048]
  const float* A    = (const float*)d_in[4];   // [4,16,2048]
  const float* Bw   = (const float*)d_in[5];   // [4,2048,16]
  float* out = (float*)d_out;

  // workspace layout (bf16):  Wb 8 MB | Ab 256 KB | Bb 256 KB | Hm 2 MB
  char* ws = (char*)d_ws;
  bf16_t* Wb = (bf16_t*)(ws);
  bf16_t* Ab = (bf16_t*)(ws + 8388608);
  bf16_t* Bb = (bf16_t*)(ws + 8650752);
  bf16_t* Hm = (bf16_t*)(ws + 8912896);

  lora_prep_kernel<<<4096, 256, 0, stream>>>(W, A, Bw, Wb, Ab, Bb);
  lora_h_kernel<<<128, 256, 0, stream>>>(x, Ab, mask, Hm);
  dim3 grid(N_OUT / 128, M_TOT / 128);
  lora_main_gemm<<<grid, 256, 0, stream>>>(x, Wb, Hm, Bb, bias, out);
}
